// Swinblock_59674275611078
// MI455X (gfx1250) — matrix operand order repaired and checked by reading, not compiled, not run on hardware
//
#include <hip/hip_runtime.h>
#include <hip/hip_bf16.h>
#include <math.h>

// ---------------------------------------------------------------------------
// Swin block on gfx1250: f16 WMMA (16x16x32) GEMMs, fused windowed attention.
// Round 4: fix async-LDS builtin pointer types (v4i* in AS1/AS3).
// ---------------------------------------------------------------------------

typedef __attribute__((ext_vector_type(16))) _Float16 v16h;
typedef __attribute__((ext_vector_type(8)))  float    v8f;
typedef __attribute__((ext_vector_type(8)))  _Float16 f16x8;
typedef __attribute__((ext_vector_type(4)))  int      v4i;

#define AS1 __attribute__((address_space(1)))
#define AS3 __attribute__((address_space(3)))

#define HH 56
#define WWID 56
#define WS 7
#define SS 3
#define NHEADS 12
#define DMODEL 384
#define HD 32
#define LW 49           // window length 7*7
#define NWIN 64         // windows per image
#define BB 32
#define LL (HH*WWID)    // 3136
#define BNW (BB*NWIN)   // 2048
#define MTOT (BNW*LW)   // 100352 rows
#define LN_EPS 1e-3f

__device__ __forceinline__ float gelu_exact(float x){
    return 0.5f * x * (1.0f + erff(x * 0.70710678118654752440f));
}

// Async 16-byte global -> LDS copy (CDNA5 GLOBAL_LOAD_ASYNC_TO_LDS_B128,
// tracked by ASYNCcnt). Fallback: plain load+store.
__device__ __forceinline__ void async_copy_b128(const void* gp, void* lp){
#if __has_builtin(__builtin_amdgcn_global_load_async_to_lds_b128)
    __builtin_amdgcn_global_load_async_to_lds_b128(
        (AS1 v4i*)gp, (AS3 v4i*)lp, 0, 0);
#else
    *(f16x8*)lp = *(const f16x8*)gp;
#endif
}
__device__ __forceinline__ void wait_async(){
#if __has_builtin(__builtin_amdgcn_global_load_async_to_lds_b128)
#  if __has_builtin(__builtin_amdgcn_s_wait_asynccnt)
    __builtin_amdgcn_s_wait_asynccnt(0);
#  else
    asm volatile("s_wait_asynccnt 0" ::: "memory");
#  endif
#endif
}

// region id for the shifted-window attention mask (replicates _make_mask).
__device__ __forceinline__ int region_id(int i, int j){
    int rg = (i < WS) ? 0 : (i < WS + SS) ? 1 : (i >= HH - SS) ? 2 : -1;
    int cg = (j < WS) ? 0 : (j < WS + SS) ? 1 : (j >= WWID - SS) ? 2 : -1;
    if (rg < 0 || cg < 0) return 0;
    return rg * 3 + cg;
}

// ---- WMMA fragment loaders from LDS (documented 16-bit layouts) -----------
// A-matrix 16x32 (MxK), row-major in LDS with row stride ld:
//   lane&15 = row M; kh = 8*(lane>>4); elem i -> K=kh+i, elem 8+i -> K=16+kh+i
__device__ __forceinline__ v16h load_a_frag(const _Float16* base, int ld){
    int lane = threadIdx.x & 31;
    const _Float16* row = base + (lane & 15) * ld + ((lane >> 4) << 3);
    f16x8 lo8 = *(const f16x8*)(row);
    f16x8 hi8 = *(const f16x8*)(row + 16);
    return __builtin_shufflevector(lo8, hi8,
        0,1,2,3,4,5,6,7,8,9,10,11,12,13,14,15);
}
// B-matrix 32x16 (KxN) staged transposed in LDS ([n][k], row stride ld):
//   lane&15 = column n; kh = 8*(lane>>4); elem i -> K=kh+i, elem 8+i -> K=16+kh+i
__device__ __forceinline__ v16h load_b_fragT(const _Float16* baseT, int ld){
    int lane = threadIdx.x & 31;
    const _Float16* row = baseT + (lane & 15) * ld + ((lane >> 4) << 3);
    f16x8 lo8 = *(const f16x8*)(row);
    f16x8 hi8 = *(const f16x8*)(row + 16);
    return __builtin_shufflevector(lo8, hi8,
        0,1,2,3,4,5,6,7,8,9,10,11,12,13,14,15);
}
// C/D 16x16 f32: lane&15 = column; VGPR r -> row = r + 8*(lane>>4)
__device__ __forceinline__ v8f load_c_frag(const float* base, int ld){
    int lane = threadIdx.x & 31;
    int lo = lane & 15, hi = lane >> 4;
    v8f c;
#pragma unroll
    for (int r = 0; r < 8; r++) c[r] = base[(r + hi * 8) * ld + lo];
    return c;
}

// ---------------------------------------------------------------------------
// Kernel 1: LayerNorm + roll(-3,-3) + window partition -> f16 [MTOT, 384]
// ---------------------------------------------------------------------------
__global__ __launch_bounds__(256) void ln_shift_window_kernel(
    const float* __restrict__ x, const float* __restrict__ gamma,
    const float* __restrict__ beta, _Float16* __restrict__ h16)
{
    int wv   = threadIdx.x >> 5;
    int lane = threadIdx.x & 31;
    int orow = blockIdx.x * 8 + wv;
    if (orow >= MTOT) return;
    int bIdx = orow / LW, pix = orow % LW;
    int b  = bIdx >> 6, w = bIdx & 63;
    int i2 = (w >> 3) * WS + pix / WS + SS; if (i2 >= HH)   i2 -= HH;
    int j2 = (w & 7)  * WS + pix % WS + SS; if (j2 >= WWID) j2 -= WWID;
    const float* row = x + ((size_t)b * LL + (size_t)i2 * WWID + j2) * DMODEL;

    float vals[12]; float s = 0.f;
#pragma unroll
    for (int t = 0; t < 12; t++){ vals[t] = row[lane + t * 32]; s += vals[t]; }
#pragma unroll
    for (int o = 16; o > 0; o >>= 1) s += __shfl_xor(s, o, 32);
    float mu = s * (1.0f / DMODEL);
    float vsum = 0.f;
#pragma unroll
    for (int t = 0; t < 12; t++){ float d = vals[t] - mu; vsum += d * d; }
#pragma unroll
    for (int o = 16; o > 0; o >>= 1) vsum += __shfl_xor(vsum, o, 32);
    float inv = rsqrtf(vsum * (1.0f / DMODEL) + LN_EPS);

    _Float16* out = h16 + (size_t)orow * DMODEL;
#pragma unroll
    for (int t = 0; t < 12; t++){
        int d = lane + t * 32;
        out[d] = (_Float16)((vals[t] - mu) * inv * gamma[d] + beta[d]);
    }
}

// Kernel 1b: plain LayerNorm (for the MLP branch), identity row mapping.
__global__ __launch_bounds__(256) void ln_plain_kernel(
    const float* __restrict__ xin, const float* __restrict__ gamma,
    const float* __restrict__ beta, _Float16* __restrict__ h16)
{
    int wv   = threadIdx.x >> 5;
    int lane = threadIdx.x & 31;
    int orow = blockIdx.x * 8 + wv;
    if (orow >= MTOT) return;
    const float* row = xin + (size_t)orow * DMODEL;
    float vals[12]; float s = 0.f;
#pragma unroll
    for (int t = 0; t < 12; t++){ vals[t] = row[lane + t * 32]; s += vals[t]; }
#pragma unroll
    for (int o = 16; o > 0; o >>= 1) s += __shfl_xor(s, o, 32);
    float mu = s * (1.0f / DMODEL);
    float vsum = 0.f;
#pragma unroll
    for (int t = 0; t < 12; t++){ float d = vals[t] - mu; vsum += d * d; }
#pragma unroll
    for (int o = 16; o > 0; o >>= 1) vsum += __shfl_xor(vsum, o, 32);
    float inv = rsqrtf(vsum * (1.0f / DMODEL) + LN_EPS);
    _Float16* out = h16 + (size_t)orow * DMODEL;
#pragma unroll
    for (int t = 0; t < 12; t++){
        int d = lane + t * 32;
        out[d] = (_Float16)((vals[t] - mu) * inv * gamma[d] + beta[d]);
    }
}

// Kernel 2: f32 [K,N] -> f16 transposed [N,K] (weights, done once per call)
__global__ void convert_transpose_kernel(const float* __restrict__ in,
                                         _Float16* __restrict__ out,
                                         int K, int N){
    int i = blockIdx.x * 256 + threadIdx.x;
    if (i < K * N){
        int k = i / N, n = i % N;
        out[(size_t)n * K + k] = (_Float16)in[i];
    }
}

// ---------------------------------------------------------------------------
// Kernel 3: tiled WMMA GEMM, C = epilogue(A[M,K] @ W[K,N] + bias).
// W supplied transposed (WT[N][K]).  256 threads = 8 waves; macro tile
// 128x128; wave = 2x4 of 16x16 tiles; BK=32; async-LDS panel staging.
// MODE 0: gelu -> f16 out (identity rows)                       [qkv, fc1]
// MODE 2: gelu + window-reverse/unshift + residual -> f32 out   [attn proj]
// MODE 3: bias-add, accumulate into f32 out                     [fc2]
// ---------------------------------------------------------------------------
template<int MODE>
__global__ __launch_bounds__(256) void gemm_kernel(
    const _Float16* __restrict__ A, const _Float16* __restrict__ WT,
    const float* __restrict__ bias, _Float16* __restrict__ out16,
    float* __restrict__ outf, const float* __restrict__ resid,
    int M, int N, int K)
{
    __shared__ __align__(16) _Float16 As[128 * 32];   // [m][k]
    __shared__ __align__(16) _Float16 Bs[128 * 32];   // [n][k]  (B transposed)

    int nb = N >> 7;
    int m0 = (blockIdx.x / nb) * 128;
    int n0 = (blockIdx.x % nb) * 128;
    int tid  = threadIdx.x;
    int wv   = tid >> 5;
    int wrow = wv >> 1, wcol = wv & 1;

    // per-thread staging coordinates: 512 chunks of 8 halfs per panel
    int ch0 = tid * 2;
    int rrA0 = (ch0    ) >> 2, ccA0 = ((ch0    ) & 3) << 3;
    int rrA1 = (ch0 + 1) >> 2, ccA1 = ((ch0 + 1) & 3) << 3;

    v8f acc[2][4];
#pragma unroll
    for (int i = 0; i < 2; i++)
#pragma unroll
        for (int j = 0; j < 4; j++) acc[i][j] = (v8f){};

    for (int k0 = 0; k0 < K; k0 += 32){
        async_copy_b128(&A [(size_t)(m0 + rrA0) * K + k0 + ccA0], &As[rrA0 * 32 + ccA0]);
        async_copy_b128(&A [(size_t)(m0 + rrA1) * K + k0 + ccA1], &As[rrA1 * 32 + ccA1]);
        async_copy_b128(&WT[(size_t)(n0 + rrA0) * K + k0 + ccA0], &Bs[rrA0 * 32 + ccA0]);
        async_copy_b128(&WT[(size_t)(n0 + rrA1) * K + k0 + ccA1], &Bs[rrA1 * 32 + ccA1]);
        if (k0 + 32 < K){   // prefetch next K panel (global_prefetch_b8)
            __builtin_prefetch(&A [(size_t)(m0 + rrA0) * K + k0 + 32 + ccA0], 0, 1);
            __builtin_prefetch(&WT[(size_t)(n0 + rrA0) * K + k0 + 32 + ccA0], 0, 1);
        }
        wait_async();
        __syncthreads();

        v16h af[2], bf[4];
#pragma unroll
        for (int i = 0; i < 2; i++)
            af[i] = load_a_frag(&As[(wrow * 32 + i * 16) * 32], 32);
#pragma unroll
        for (int j = 0; j < 4; j++)
            bf[j] = load_b_fragT(&Bs[(wcol * 64 + j * 16) * 32], 32);
#pragma unroll
        for (int i = 0; i < 2; i++)
#pragma unroll
            for (int j = 0; j < 4; j++)
                acc[i][j] = __builtin_amdgcn_wmma_f32_16x16x32_f16(
                    false, af[i], false, bf[j], (short)0, acc[i][j], false, false);
        __syncthreads();
    }

    int lane = tid & 31;
    int hi = lane >> 4, lo = lane & 15;
#pragma unroll
    for (int i = 0; i < 2; i++){
#pragma unroll
        for (int j = 0; j < 4; j++){
            int col = n0 + wcol * 64 + j * 16 + lo;
            float bval = bias[col];
#pragma unroll
            for (int r = 0; r < 8; r++){
                int row = m0 + wrow * 32 + i * 16 + r + hi * 8;
                float v = acc[i][j][r] + bval;
                if (MODE == 0){
                    out16[(size_t)row * N + col] = (_Float16)gelu_exact(v);
                } else if (MODE == 2){
                    // window reverse + roll(+3,+3) + residual add
                    int bIdx = row / LW, pix = row % LW;
                    int b  = bIdx >> 6, w = bIdx & 63;
                    int ii = (w >> 3) * WS + pix / WS + SS; if (ii >= HH)   ii -= HH;
                    int jj = (w & 7)  * WS + pix % WS + SS; if (jj >= WWID) jj -= WWID;
                    size_t srow = (size_t)b * LL + (size_t)ii * WWID + jj;
                    outf[srow * DMODEL + col] =
                        resid[srow * DMODEL + col] + gelu_exact(v);
                } else { // MODE 3
                    outf[(size_t)row * DMODEL + col] += v;
                }
            }
        }
    }
}

// ---------------------------------------------------------------------------
// Kernel 4: fused windowed attention, one block per (window-batch b, head h).
// Bias+mask table built in LDS, fed to WMMA as C operand.  q/k tiles staged
// with async-LDS loads.  S = qk^T + C, softmax, O = P V.
// ---------------------------------------------------------------------------
__global__ __launch_bounds__(128) void attention_kernel(
    const _Float16* __restrict__ qkv,        // [MTOT, 3*DMODEL]
    const float* __restrict__ bias_table,    // [313, NHEADS]
    _Float16* __restrict__ out16)            // [MTOT, DMODEL]
{
    __shared__ __align__(16) _Float16 qs[64 * 32];   // [l][d]
    __shared__ __align__(16) _Float16 ks[64 * 32];   // [l][d] == B^T for qk^T
    __shared__ __align__(16) _Float16 vt[32 * 64];   // [d][l] == B^T for P@V
    __shared__ __align__(16) float    Ss[64 * 64];
    __shared__ __align__(16) _Float16 Ps[64 * 64];

    int bh = blockIdx.x;
    int b  = bh / NHEADS;
    int h  = bh % NHEADS;
    int wwin = b & 63;
    int wr = wwin >> 3, wc = wwin & 7;
    int tid = threadIdx.x;

    // stage q, k via async b128 chunks (8 halfs); zero-pad rows >= LW
    for (int c = tid; c < 256; c += 128){
        int l = c >> 2, d8 = (c & 3) << 3;
        if (l < LW){
            const _Float16* src = qkv + ((size_t)b * LW + l) * (3 * DMODEL) + h * HD + d8;
            async_copy_b128(src,          &qs[l * 32 + d8]);
            async_copy_b128(src + DMODEL, &ks[l * 32 + d8]);
        } else {
            *(f16x8*)&qs[l * 32 + d8] = (f16x8){};
            *(f16x8*)&ks[l * 32 + d8] = (f16x8){};
        }
    }
    // stage v transposed ([d][l]); zero-pad cols >= LW
    for (int e = tid; e < 64 * 32; e += 128){
        int l = e >> 5, d = e & 31;
        _Float16 vv = (_Float16)0.f;
        if (l < LW)
            vv = qkv[((size_t)b * LW + l) * (3 * DMODEL) + 2 * DMODEL + h * HD + d];
        vt[d * 64 + l] = vv;
    }
    // build additive table (rel-pos bias + shift mask) into Ss
    for (int e = tid; e < 64 * 64; e += 128){
        int l = e >> 6, m = e & 63;
        float v = 0.f;
        if (l < LW && m < LW){
            int rel = 13 * ((m % 7 - l % 7 + 6) + (m / 7 - l / 7 + 6));
            v = bias_table[rel * NHEADS + h];
            int idl = region_id(wr * WS + l / 7, wc * WS + l % 7);
            int idm = region_id(wr * WS + m / 7, wc * WS + m % 7);
            if (idl != idm) v -= 100.0f;
        }
        Ss[e] = v;
    }
    wait_async();
    __syncthreads();

    int wv = tid >> 5;
    int lane = tid & 31, hi = lane >> 4, lo = lane & 15;

    // ---- S = q @ k^T + table (wave wv owns row-tile wv; 4 column tiles) ----
    {
        v16h af = load_a_frag(&qs[(wv * 16) * 32], 32);
#pragma unroll
        for (int nt = 0; nt < 4; nt++){
            v16h bf = load_b_fragT(&ks[(nt * 16) * 32], 32);
            v8f  cf = load_c_frag(&Ss[(wv * 16) * 64 + nt * 16], 64);
            v8f acc = __builtin_amdgcn_wmma_f32_16x16x32_f16(
                false, af, false, bf, (short)0, cf, false, false);
#pragma unroll
            for (int r = 0; r < 8; r++)
                Ss[(wv * 16 + r + hi * 8) * 64 + nt * 16 + lo] = acc[r];
        }
    }
    __syncthreads();

    // ---- softmax over valid 49 columns; zero-pad P elsewhere ----
    if (tid < 64){
        int row = tid;
        if (row < LW){
            float mx = -1e30f;
            for (int c2 = 0; c2 < LW; c2++) mx = fmaxf(mx, Ss[row * 64 + c2]);
            float sum = 0.f;
            for (int c2 = 0; c2 < LW; c2++){
                float e = __expf(Ss[row * 64 + c2] - mx);
                Ss[row * 64 + c2] = e; sum += e;
            }
            float rinv = 1.0f / sum;
            for (int c2 = 0; c2 < LW; c2++)
                Ps[row * 64 + c2] = (_Float16)(Ss[row * 64 + c2] * rinv);
            for (int c2 = LW; c2 < 64; c2++) Ps[row * 64 + c2] = (_Float16)0.f;
        } else {
            for (int c2 = 0; c2 < 64; c2++) Ps[row * 64 + c2] = (_Float16)0.f;
        }
    }
    __syncthreads();

    // ---- O = P @ V (M=64 pad, K=64 pad, N=32) ----
#pragma unroll
    for (int nt = 0; nt < 2; nt++){
        v8f acc = (v8f){};
#pragma unroll
        for (int s2 = 0; s2 < 2; s2++){
            v16h af = load_a_frag(&Ps[(wv * 16) * 64 + s2 * 32], 64);
            v16h bf = load_b_fragT(&vt[(nt * 16) * 64 + s2 * 32], 64);
            acc = __builtin_amdgcn_wmma_f32_16x16x32_f16(
                false, af, false, bf, (short)0, acc, false, false);
        }
        int col = nt * 16 + lo;
#pragma unroll
        for (int r = 0; r < 8; r++){
            int row = wv * 16 + r + hi * 8;
            if (row < LW)
                out16[((size_t)b * LW + row) * DMODEL + h * HD + col] = (_Float16)acc[r];
        }
    }
}

// ---------------------------------------------------------------------------
static inline size_t alignup(size_t v){ return (v + 255) & ~(size_t)255; }

extern "C" void kernel_launch(void* const* d_in, const int* in_sizes, int n_in,
                              void* d_out, int out_size, void* d_ws, size_t ws_size,
                              hipStream_t stream) {
    (void)in_sizes; (void)n_in; (void)out_size; (void)ws_size;
    const float* x          = (const float*)d_in[0];
    const float* gamma      = (const float*)d_in[1];
    const float* beta       = (const float*)d_in[2];
    const float* w_qkv      = (const float*)d_in[3];
    const float* b_qkv      = (const float*)d_in[4];
    const float* bias_table = (const float*)d_in[5];
    const float* w_att      = (const float*)d_in[6];
    const float* b_att      = (const float*)d_in[7];
    const float* w_fc1      = (const float*)d_in[8];
    const float* b_fc1      = (const float*)d_in[9];
    const float* w_fc2      = (const float*)d_in[10];
    const float* b_fc2      = (const float*)d_in[11];
    float* out = (float*)d_out;

    char* ws = (char*)d_ws;
    size_t off = 0;
    _Float16* wqkvT = (_Float16*)(ws + off); off = alignup(off + (size_t)DMODEL * 3 * DMODEL * 2);
    _Float16* wattT = (_Float16*)(ws + off); off = alignup(off + (size_t)DMODEL * DMODEL * 2);
    _Float16* wfc1T = (_Float16*)(ws + off); off = alignup(off + (size_t)DMODEL * 4 * DMODEL * 2);
    _Float16* wfc2T = (_Float16*)(ws + off); off = alignup(off + (size_t)4 * DMODEL * DMODEL * 2);
    // big region: holds h16 [MTOT,384] + qkv16 [MTOT,1152]; later reused as
    // fc1 output [MTOT,1536] (sizes match exactly).
    _Float16* big16  = (_Float16*)(ws + off); off = alignup(off + (size_t)MTOT * 1536 * 2);
    _Float16* h16    = big16;
    _Float16* qkv16  = big16 + (size_t)MTOT * DMODEL;
    _Float16* fc116  = big16;
    // att16: attention output [MTOT,384]; later reused for h2 (second LN).
    _Float16* att16  = (_Float16*)(ws + off); off = alignup(off + (size_t)MTOT * DMODEL * 2);
    _Float16* h216   = att16;

    // 1) weight conversions (f32 [K,N] -> f16 [N,K])
    {
        int n;
        n = DMODEL * 3 * DMODEL; convert_transpose_kernel<<<(n + 255) / 256, 256, 0, stream>>>(w_qkv, wqkvT, DMODEL, 3 * DMODEL);
        n = DMODEL * DMODEL;     convert_transpose_kernel<<<(n + 255) / 256, 256, 0, stream>>>(w_att, wattT, DMODEL, DMODEL);
        n = DMODEL * 4 * DMODEL; convert_transpose_kernel<<<(n + 255) / 256, 256, 0, stream>>>(w_fc1, wfc1T, DMODEL, 4 * DMODEL);
        n = 4 * DMODEL * DMODEL; convert_transpose_kernel<<<(n + 255) / 256, 256, 0, stream>>>(w_fc2, wfc2T, 4 * DMODEL, DMODEL);
    }

    // 2) LN + shift + window partition
    ln_shift_window_kernel<<<MTOT / 8, 256, 0, stream>>>(x, gamma, beta, h16);

    // 3) qkv projection + GELU  (M=100352, N=1152, K=384)
    gemm_kernel<0><<<(MTOT / 128) * (1152 / 128), 256, 0, stream>>>(
        h16, wqkvT, b_qkv, qkv16, nullptr, nullptr, MTOT, 3 * DMODEL, DMODEL);

    // 4) windowed attention (bias + mask + softmax fused)
    attention_kernel<<<BNW * NHEADS, 128, 0, stream>>>(qkv16, bias_table, att16);

    // 5) attention output projection + GELU + window-reverse/unshift + residual
    gemm_kernel<2><<<(MTOT / 128) * (DMODEL / 128), 256, 0, stream>>>(
        att16, wattT, b_att, nullptr, out, x, MTOT, DMODEL, DMODEL);

    // 6) second LN (shared params), plain layout
    ln_plain_kernel<<<MTOT / 8, 256, 0, stream>>>(out, gamma, beta, h216);

    // 7) fc1 + GELU  (M=100352, N=1536, K=384)
    gemm_kernel<0><<<(MTOT / 128) * (1536 / 128), 256, 0, stream>>>(
        h216, wfc1T, b_fc1, fc116, nullptr, nullptr, MTOT, 4 * DMODEL, DMODEL);

    // 8) fc2, accumulate into d_out  (M=100352, N=384, K=1536)
    gemm_kernel<3><<<(MTOT / 128) * (DMODEL / 128), 256, 0, stream>>>(
        fc116, wfc2T, b_fc2, nullptr, out, nullptr, MTOT, DMODEL, 4 * DMODEL);
}
